// SessionGraph_52871047414133
// MI455X (gfx1250) — matrix operand order repaired and checked by reading, not compiled, not run on hardware
//
#include <hip/hip_runtime.h>
#include <hip/hip_bf16.h>
#include <math.h>
#include <stdint.h>

typedef _Float16 h16;
typedef __attribute__((ext_vector_type(16))) _Float16 v16h;
typedef __attribute__((ext_vector_type(8)))  _Float16 v8h;
typedef __attribute__((ext_vector_type(8)))  float    v8f;

#define BS 64
#define SL 50
#define HS 256
#define NN 40000

// --- CDNA5 async global->LDS copy (ASYNCcnt-tracked), with sync fallback ----
#if defined(__has_builtin)
#if __has_builtin(__builtin_amdgcn_global_load_async_to_lds_b128)
#define USE_ASYNC_LDS 1
#endif
#endif

#ifdef USE_ASYNC_LDS
typedef int v4i_raw __attribute__((__vector_size__(16)));
typedef __attribute__((address_space(1))) v4i_raw* glb_v4i_p;
typedef __attribute__((address_space(3))) v4i_raw* lds_v4i_p;

__device__ __forceinline__ void async_copy_b128(const void* g, void* l) {
  __builtin_amdgcn_global_load_async_to_lds_b128(
      (glb_v4i_p)(uintptr_t)g, (lds_v4i_p)(uintptr_t)l, 0, 0);
}
__device__ __forceinline__ void wait_async_lds() {
#if __has_builtin(__builtin_amdgcn_s_wait_asynccnt)
  __builtin_amdgcn_s_wait_asynccnt(0);
#else
  asm volatile("s_wait_asynccnt 0" ::: "memory");
#endif
}
#endif

__device__ __forceinline__ v16h cat8(v8h lo, v8h hi) {
  return __builtin_shufflevector(lo, hi, 0,1,2,3,4,5,6,7,8,9,10,11,12,13,14,15);
}
__device__ __forceinline__ v8f wmma16(v16h a, v16h b, v8f c) {
  // D = A(16x32) * B(32x16) + C, f32 accumulate
  return __builtin_amdgcn_wmma_f32_16x16x32_f16(false, a, false, b, (short)0, c, false, false);
}

// ---------------------------------------------------------------------------
// Generic linear: out[M,N] = A[M,K](f16) x W[N,K](f16)^T + bias (nn.Linear)
// Block tile 64(M) x 128(N), 8 waves as 2x4, each wave 32x32 (2x2 wmma accs).
// Interior blocks use GLOBAL_LOAD_ASYNC_TO_LDS_B128 (no VGPR staging).
// ---------------------------------------------------------------------------
__global__ __launch_bounds__(256)
void wmma_linear(const h16* __restrict__ A, const h16* __restrict__ W,
                 const float* __restrict__ bias,
                 h16* __restrict__ outH, float* __restrict__ outF,
                 int M, int N, int K)
{
  __shared__ h16 sA[64 * 56];
  __shared__ h16 sB[128 * 56];
  const int tid  = threadIdx.x;
  const int lane = tid & 31;
  const int w    = tid >> 5;
  const int wm   = w & 1;     // 2 M-groups of 32 rows
  const int wn   = w >> 1;    // 4 N-groups of 32 cols
  const int m0   = blockIdx.x * 64;
  const int n0   = blockIdx.y * 128;
  const bool interior = (m0 + 64 <= M) && (n0 + 128 <= N);

  const int rowA = tid >> 2, colA = (tid & 3) * 8;   // A tile 64x32
  const int rowB = tid >> 1, colB = (tid & 1) * 16;  // B tile 128x32

  v8f acc[2][2] = {};

  for (int k0 = 0; k0 < K; k0 += 32) {
    if (interior) {
      const h16* ga = A + (size_t)(m0 + rowA) * K + k0 + colA;
      const h16* gb = W + (size_t)(n0 + rowB) * K + k0 + colB;
#ifdef USE_ASYNC_LDS
      async_copy_b128(ga,     sA + rowA * 56 + colA);
      async_copy_b128(gb,     sB + rowB * 56 + colB);
      async_copy_b128(gb + 8, sB + rowB * 56 + colB + 8);
      wait_async_lds();
#else
      *(v8h*)(sA + rowA * 56 + colA)     = *(const v8h*)(ga);
      *(v8h*)(sB + rowB * 56 + colB)     = *(const v8h*)(gb);
      *(v8h*)(sB + rowB * 56 + colB + 8) = *(const v8h*)(gb + 8);
#endif
      if (k0 + 32 < K) {
        __builtin_prefetch(ga + 32, 0, 1);
        __builtin_prefetch(gb + 32, 0, 1);
      }
    } else {
      int gm = m0 + rowA;
      v8h v = {};
      if (gm < M) v = *(const v8h*)(A + (size_t)gm * K + k0 + colA);
      *(v8h*)(sA + rowA * 56 + colA) = v;
      int gn = n0 + rowB;
      v8h v0 = {}, v1 = {};
      if (gn < N) {
        const h16* p = W + (size_t)gn * K + k0 + colB;
        v0 = *(const v8h*)(p);
        v1 = *(const v8h*)(p + 8);
      }
      *(v8h*)(sB + rowB * 56 + colB)     = v0;
      *(v8h*)(sB + rowB * 56 + colB + 8) = v1;
    }
    __syncthreads();

    const int kaA = (lane < 16) ? 0 : 8;    // A: K 0-7/16-23 vs 8-15/24-31
    const int kaB = (lane < 16) ? 0 : 8;    // B: K 0-7/16-23 vs 8-15/24-31
    v16h af[2], bf[2];
#pragma unroll
    for (int mi = 0; mi < 2; ++mi) {
      int r = wm * 32 + mi * 16 + (lane & 15);
      v8h lo = *(const v8h*)(sA + r * 56 + kaA);
      v8h hi = *(const v8h*)(sA + r * 56 + kaA + 16);
      af[mi] = cat8(lo, hi);
    }
#pragma unroll
    for (int ni = 0; ni < 2; ++ni) {
      int r = wn * 32 + ni * 16 + (lane & 15);
      v8h lo = *(const v8h*)(sB + r * 56 + kaB);
      v8h hi = *(const v8h*)(sB + r * 56 + kaB + 16);
      bf[ni] = cat8(lo, hi);
    }
#pragma unroll
    for (int mi = 0; mi < 2; ++mi)
#pragma unroll
      for (int ni = 0; ni < 2; ++ni)
        acc[mi][ni] = wmma16(af[mi], bf[ni], acc[mi][ni]);
    __syncthreads();
  }

  const int cn    = lane & 15;
  const int rbase = (lane >> 4) * 8;
#pragma unroll
  for (int mi = 0; mi < 2; ++mi)
#pragma unroll
    for (int ni = 0; ni < 2; ++ni) {
      int gcol = n0 + wn * 32 + ni * 16 + cn;
      float bv = (bias != nullptr && gcol < N) ? bias[gcol] : 0.f;
#pragma unroll
      for (int j = 0; j < 8; ++j) {
        int grow = m0 + wm * 32 + mi * 16 + rbase + j;
        if (grow < M && gcol < N) {
          float v = acc[mi][ni][j] + bv;
          if (outF) outF[(size_t)grow * N + gcol] = v;
          if (outH) outH[(size_t)grow * N + gcol] = (h16)v;
        }
      }
    }
}

// ---------------------------------------------------------------------------
// Batched attention scores: S[batch, 50, 50] = Q[batch] x K[batch]^T
// Q,K: [batch][50][256] f16. One block per batch, 64x64 padded tile.
// M/N pad rows are garbage (clamped row load) -- never stored; K is exact.
// ---------------------------------------------------------------------------
__global__ __launch_bounds__(256)
void wmma_scores(const h16* __restrict__ Qb, const h16* __restrict__ Kb,
                 float* __restrict__ S)
{
  __shared__ h16 sA[64 * 56];
  __shared__ h16 sB[64 * 56];
  const int batch = blockIdx.x;
  const h16* Aq = Qb + (size_t)batch * (SL * HS);
  const h16* Bk = Kb + (size_t)batch * (SL * HS);
  const int tid = threadIdx.x, lane = tid & 31, w = tid >> 5;
  const int wm = w & 1, wn = w >> 1;   // 2x32 rows, 4x16 cols

  const int row = tid >> 2, col = (tid & 3) * 8;
  const int crow = (row < SL) ? row : (SL - 1);     // clamp: pad rows unused

  v8f acc[2] = {};
  for (int k0 = 0; k0 < HS; k0 += 32) {
#ifdef USE_ASYNC_LDS
    async_copy_b128(Aq + (size_t)crow * HS + k0 + col, sA + row * 56 + col);
    async_copy_b128(Bk + (size_t)crow * HS + k0 + col, sB + row * 56 + col);
    wait_async_lds();
#else
    *(v8h*)(sA + row * 56 + col) = *(const v8h*)(Aq + (size_t)crow * HS + k0 + col);
    *(v8h*)(sB + row * 56 + col) = *(const v8h*)(Bk + (size_t)crow * HS + k0 + col);
#endif
    __syncthreads();

    const int kaA = (lane < 16) ? 0 : 8;
    const int kaB = (lane < 16) ? 0 : 8;
    v16h af[2], bf;
#pragma unroll
    for (int mi = 0; mi < 2; ++mi) {
      int r = wm * 32 + mi * 16 + (lane & 15);
      af[mi] = cat8(*(const v8h*)(sA + r * 56 + kaA),
                    *(const v8h*)(sA + r * 56 + kaA + 16));
    }
    {
      int r = wn * 16 + (lane & 15);
      bf = cat8(*(const v8h*)(sB + r * 56 + kaB),
                *(const v8h*)(sB + r * 56 + kaB + 16));
    }
#pragma unroll
    for (int mi = 0; mi < 2; ++mi) acc[mi] = wmma16(af[mi], bf, acc[mi]);
    __syncthreads();
  }

  const int cn = lane & 15, rbase = (lane >> 4) * 8;
#pragma unroll
  for (int mi = 0; mi < 2; ++mi) {
    int gcol = wn * 16 + cn;
#pragma unroll
    for (int j = 0; j < 8; ++j) {
      int grow = wm * 32 + mi * 16 + rbase + j;
      if (grow < SL && gcol < SL)
        S[(size_t)batch * (SL * SL) + grow * SL + gcol] = acc[mi][j];
    }
  }
}

// ---------------------------------------------------------------------------
// Fused scale + mask + softmax over last axis (50). One wave per row.
// simBroadcastR: sim indexed [b,q,k] (stage 1) vs [b,r,q,k] (stage 2).
// ---------------------------------------------------------------------------
__global__ __launch_bounds__(256)
void softmax_rows(const float* __restrict__ S, const float* __restrict__ mask,
                  h16* __restrict__ P, float scale, int simBroadcastR)
{
  const int row  = blockIdx.x * 8 + (threadIdx.x >> 5);   // (b*50+r)*50+q
  const int lane = threadIdx.x & 31;
  const int b = row / (SL * SL);
  const int rq = row % (SL * SL);
  const int q = rq % SL;
  const float* sp = simBroadcastR ? (S + ((size_t)(b * SL + q)) * SL)
                                  : (S + (size_t)row * SL);
  const float* mp = mask + (size_t)row * SL;

  float x0 = sp[lane] * scale + mp[lane];                       // lane < 32 < 50
  float x1 = (lane + 32 < SL) ? (sp[lane + 32] * scale + mp[lane + 32]) : -1e30f;
  float m = fmaxf(x0, x1);
#pragma unroll
  for (int o = 16; o; o >>= 1) m = fmaxf(m, __shfl_xor(m, o, 32));
  float e0 = __expf(x0 - m);
  float e1 = (lane + 32 < SL) ? __expf(x1 - m) : 0.f;
  float s = e0 + e1;
#pragma unroll
  for (int o = 16; o; o >>= 1) s += __shfl_xor(s, o, 32);
  float inv = 1.f / s;
  P[(size_t)row * SL + lane] = (h16)(e0 * inv);
  if (lane + 32 < SL) P[(size_t)row * SL + lane + 32] = (h16)(e1 * inv);
}

// ---------------------------------------------------------------------------
// Batched apply: for batch br=(b*50+r): Out[b,q,r,:] = P[br] x V (store
// TRANSPOSED so the reference .transpose(0,2,1,3) is free).
// V = Vbase + (vPerBatch ? br : b)*50*256.  K padded 50->64 (zeros).
// ---------------------------------------------------------------------------
__global__ __launch_bounds__(256)
void wmma_apply(const h16* __restrict__ P, const h16* __restrict__ Vbase,
                h16* __restrict__ Out, int vPerBatch)
{
  __shared__ h16 sP[64 * 72];     // [q][k]
  __shared__ h16 sVT[256 * 72];   // [d][k] (transposed V)
  const int br = blockIdx.x;
  const int b = br / SL, r = br % SL;
  const h16* Pp = P + (size_t)br * (SL * SL);
  const h16* V  = Vbase + (size_t)(vPerBatch ? br : b) * (SL * HS);
  const int tid = threadIdx.x, lane = tid & 31, w = tid >> 5;
  const int wm = w & 1, wn = w >> 1;   // 2x32 rows(q), 4x64 cols(d)

  for (int i = tid; i < 64 * 64; i += 256) {
    int q = i >> 6, k = i & 63;
    sP[q * 72 + k] = (q < SL && k < SL) ? Pp[q * SL + k] : (h16)0;
  }
  for (int i = tid; i < 64 * 256; i += 256) {
    int k = i >> 8, d = i & 255;
    sVT[d * 72 + k] = (k < SL) ? V[(size_t)k * HS + d] : (h16)0;
  }
  __syncthreads();

  v8f acc[2][4] = {};
#pragma unroll
  for (int k0 = 0; k0 < 64; k0 += 32) {
    const int kaA = k0 + ((lane < 16) ? 0 : 8);
    const int kaB = k0 + ((lane < 16) ? 0 : 8);
    v16h af[2], bf[4];
#pragma unroll
    for (int mi = 0; mi < 2; ++mi) {
      int rr = wm * 32 + mi * 16 + (lane & 15);
      af[mi] = cat8(*(const v8h*)(sP + rr * 72 + kaA),
                    *(const v8h*)(sP + rr * 72 + kaA + 16));
    }
#pragma unroll
    for (int ni = 0; ni < 4; ++ni) {
      int d = wn * 64 + ni * 16 + (lane & 15);
      bf[ni] = cat8(*(const v8h*)(sVT + d * 72 + kaB),
                    *(const v8h*)(sVT + d * 72 + kaB + 16));
    }
#pragma unroll
    for (int mi = 0; mi < 2; ++mi)
#pragma unroll
      for (int ni = 0; ni < 4; ++ni)
        acc[mi][ni] = wmma16(af[mi], bf[ni], acc[mi][ni]);
  }

  const int cn = lane & 15, rbase = (lane >> 4) * 8;
#pragma unroll
  for (int mi = 0; mi < 2; ++mi)
#pragma unroll
    for (int ni = 0; ni < 4; ++ni) {
      int d = wn * 64 + ni * 16 + cn;
#pragma unroll
      for (int j = 0; j < 8; ++j) {
        int q = wm * 32 + mi * 16 + rbase + j;
        if (q < SL)
          Out[(((size_t)(b * SL + q)) * SL + r) * HS + d] = (h16)acc[mi][ni][j];
      }
    }
}

// ---------------------------------------------------------------------------
// Small helper kernels
// ---------------------------------------------------------------------------
__global__ __launch_bounds__(256)
void cvt_f32_f16(const float* __restrict__ in, h16* __restrict__ out, int n)
{
  int i = blockIdx.x * 256 + threadIdx.x;
  if (i < n) out[i] = (h16)in[i];
}

__global__ __launch_bounds__(256)
void gather_h0(const int* __restrict__ inputs, const float* __restrict__ emb,
               h16* __restrict__ h0)
{
  int row = blockIdx.x, d = threadIdx.x;
  int idx = inputs[row];
  h0[(size_t)row * HS + d] = (h16)emb[(size_t)idx * HS + d];
}

__global__ __launch_bounds__(256)
void extract_diag(const h16* __restrict__ H, h16* __restrict__ out)
{
  int row = blockIdx.x;                 // b*50+i
  int b = row / SL, i = row % SL, d = threadIdx.x;
  out[(size_t)row * HS + d] = H[(((size_t)(b * SL + i)) * SL + i) * HS + d];
}

// Stage 3: only the diagonal q-row is needed per (b,i).
__global__ __launch_bounds__(256)
void stage3_diag(const h16* __restrict__ q3d, const h16* __restrict__ k3,
                 const h16* __restrict__ H2, const float* __restrict__ mask,
                 h16* __restrict__ d3, float scale)
{
  __shared__ float sS[64];
  __shared__ float sP[64];
  const int row = blockIdx.x;           // b*50+i
  const int i = row % SL;
  const int tid = threadIdx.x, lane = tid & 31, w = tid >> 5;
  const h16* qv = q3d + (size_t)row * HS;

  for (int k = w; k < SL; k += 8) {
    const h16* kv = k3 + ((size_t)row * SL + k) * HS;
    float acc = 0.f;
    for (int d = lane; d < HS; d += 32) acc += (float)qv[d] * (float)kv[d];
#pragma unroll
    for (int o = 16; o; o >>= 1) acc += __shfl_xor(acc, o, 32);
    if (lane == 0)
      sS[k] = acc * scale + mask[((size_t)row * SL + i) * SL + k];
  }
  __syncthreads();
  if (w == 0) {
    float x0 = sS[lane];
    float x1 = (lane + 32 < SL) ? sS[lane + 32] : -1e30f;
    float m = fmaxf(x0, x1);
#pragma unroll
    for (int o = 16; o; o >>= 1) m = fmaxf(m, __shfl_xor(m, o, 32));
    float e0 = __expf(x0 - m);
    float e1 = (lane + 32 < SL) ? __expf(x1 - m) : 0.f;
    float s = e0 + e1;
#pragma unroll
    for (int o = 16; o; o >>= 1) s += __shfl_xor(s, o, 32);
    float inv = 1.f / s;
    sP[lane] = e0 * inv;
    if (lane + 32 < SL) sP[lane + 32] = e1 * inv;
  }
  __syncthreads();
  int d = tid;
  float acc = 0.f;
  for (int k = 0; k < SL; ++k)
    acc += sP[k] * (float)H2[((size_t)row * SL + k) * HS + d];
  d3[(size_t)row * HS + d] = (h16)acc;
}

__global__ __launch_bounds__(256)
void build_cat(const h16* __restrict__ d1, const h16* __restrict__ d2,
               const h16* __restrict__ d3, h16* __restrict__ cat)
{
  int row = blockIdx.x, d = threadIdx.x;
  cat[(size_t)row * (3 * HS) + d]          = d1[(size_t)row * HS + d];
  cat[(size_t)row * (3 * HS) + HS + d]     = d2[(size_t)row * HS + d];
  cat[(size_t)row * (3 * HS) + 2 * HS + d] = d3[(size_t)row * HS + d];
}

__global__ __launch_bounds__(256)
void norm_rows(const float* __restrict__ in, h16* __restrict__ out)
{
  __shared__ float red[8];
  __shared__ float total;
  int row = blockIdx.x, tid = threadIdx.x, lane = tid & 31, w = tid >> 5;
  float v = in[(size_t)row * HS + tid];
  float ss = v * v;
#pragma unroll
  for (int o = 16; o; o >>= 1) ss += __shfl_xor(ss, o, 32);
  if (lane == 0) red[w] = ss;
  __syncthreads();
  if (tid == 0) {
    float t = 0.f;
#pragma unroll
    for (int j = 0; j < 8; ++j) t += red[j];
    total = t;
  }
  __syncthreads();
  out[(size_t)row * HS + tid] = (h16)(v * rsqrtf(total));
}

// ---------------------------------------------------------------------------
// Host orchestration
// ---------------------------------------------------------------------------
extern "C" void kernel_launch(void* const* d_in, const int* in_sizes, int n_in,
                              void* d_out, int out_size, void* d_ws, size_t ws_size,
                              hipStream_t stream)
{
  (void)in_sizes; (void)n_in; (void)out_size; (void)ws_size;
  const int*   inputs     = (const int*)  d_in[0];
  const float* struct_m   = (const float*)d_in[1];
  const float* temporal_m = (const float*)d_in[2];
  const float* sasrec_m   = (const float*)d_in[3];
  const float* emb        = (const float*)d_in[4];
  const float* sq_w = (const float*)d_in[5],  *sq_b = (const float*)d_in[6];
  const float* sk_w = (const float*)d_in[7],  *sk_b = (const float*)d_in[8];
  const float* tq_w = (const float*)d_in[9],  *tq_b = (const float*)d_in[10];
  const float* tk_w = (const float*)d_in[11], *tk_b = (const float*)d_in[12];
  const float* aq_w = (const float*)d_in[13], *aq_b = (const float*)d_in[14];
  const float* ak_w = (const float*)d_in[15], *ak_b = (const float*)d_in[16];
  const float* cat_w = (const float*)d_in[17], *cat_b = (const float*)d_in[18];
  float* scores_out = (float*)d_out;

  const int M1 = BS * SL;        // 3200
  const int MF = BS * SL * SL;   // 160000
  const int NM1 = NN - 1;        // 39999
  const float scale = 0.0625f;   // 1/sqrt(256)

  // bump allocator over workspace
  char* base = (char*)d_ws;
  size_t off = 0;
  auto alloc = [&](size_t bytes) -> void* {
    void* p = base + off;
    off = (off + bytes + 255) & ~(size_t)255;
    return p;
  };

  h16* wsq = (h16*)alloc((size_t)HS * HS * 2);
  h16* wsk = (h16*)alloc((size_t)HS * HS * 2);
  h16* wtq = (h16*)alloc((size_t)HS * HS * 2);
  h16* wtk = (h16*)alloc((size_t)HS * HS * 2);
  h16* waq = (h16*)alloc((size_t)HS * HS * 2);
  h16* wak = (h16*)alloc((size_t)HS * HS * 2);
  h16* wct = (h16*)alloc((size_t)HS * 3 * HS * 2);
  h16* h0h = (h16*)alloc((size_t)M1 * HS * 2);
  h16* Q0h = (h16*)alloc((size_t)M1 * HS * 2);
  h16* K0h = (h16*)alloc((size_t)M1 * HS * 2);
  float* Sbuf = (float*)alloc((size_t)MF * SL * 4);   // reused stage1 (prefix) & 2
  h16* Pbuf = (h16*)alloc((size_t)MF * SL * 2);       // reused
  h16* H1T  = (h16*)alloc((size_t)MF * HS * 2);
  h16* q2h  = (h16*)alloc((size_t)MF * HS * 2);       // reused as k3
  h16* k2h  = (h16*)alloc((size_t)MF * HS * 2);
  h16* H2   = (h16*)alloc((size_t)MF * HS * 2);
  h16* q3dh = (h16*)alloc((size_t)M1 * HS * 2);
  h16* dh1  = (h16*)alloc((size_t)M1 * HS * 2);
  h16* dh2  = (h16*)alloc((size_t)M1 * HS * 2);
  h16* d3h  = (h16*)alloc((size_t)M1 * HS * 2);
  h16* catC = (h16*)alloc((size_t)M1 * 3 * HS * 2);
  float* Af = (float*)alloc((size_t)M1 * HS * 4);
  h16* Anh  = (h16*)alloc((size_t)M1 * HS * 2);
  h16* Bnh  = (h16*)alloc((size_t)NM1 * HS * 2);
  h16* k3h  = q2h;

  const int WN = (HS * HS + 255) / 256;
  cvt_f32_f16<<<WN, 256, 0, stream>>>(sq_w, wsq, HS * HS);
  cvt_f32_f16<<<WN, 256, 0, stream>>>(sk_w, wsk, HS * HS);
  cvt_f32_f16<<<WN, 256, 0, stream>>>(tq_w, wtq, HS * HS);
  cvt_f32_f16<<<WN, 256, 0, stream>>>(tk_w, wtk, HS * HS);
  cvt_f32_f16<<<WN, 256, 0, stream>>>(aq_w, waq, HS * HS);
  cvt_f32_f16<<<WN, 256, 0, stream>>>(ak_w, wak, HS * HS);
  cvt_f32_f16<<<(HS * 3 * HS + 255) / 256, 256, 0, stream>>>(cat_w, wct, HS * 3 * HS);

  gather_h0<<<M1, 256, 0, stream>>>(inputs, emb, h0h);

  dim3 gSmall((M1 + 63) / 64, (HS + 127) / 128);      // 50 x 2
  dim3 gBig((MF + 63) / 64, (HS + 127) / 128);        // 2500 x 2

  // ---- stage 1 (Q/K independent of broadcast dim r) ----
  wmma_linear<<<gSmall, 256, 0, stream>>>(h0h, wsq, sq_b, Q0h, nullptr, M1, HS, HS);
  wmma_linear<<<gSmall, 256, 0, stream>>>(h0h, wsk, sk_b, K0h, nullptr, M1, HS, HS);
  wmma_scores<<<BS, 256, 0, stream>>>(Q0h, K0h, Sbuf);
  softmax_rows<<<MF / 8, 256, 0, stream>>>(Sbuf, struct_m, Pbuf, scale, 1);
  wmma_apply<<<M1, 256, 0, stream>>>(Pbuf, h0h, H1T, 0);   // H1T = ht (transposed)
  extract_diag<<<M1, 256, 0, stream>>>(H1T, dh1);          // d1

  // ---- stage 2 (no scale) ----
  wmma_linear<<<gBig, 256, 0, stream>>>(H1T, wtq, tq_b, q2h, nullptr, MF, HS, HS);
  wmma_linear<<<gBig, 256, 0, stream>>>(H1T, wtk, tk_b, k2h, nullptr, MF, HS, HS);
  wmma_scores<<<M1, 256, 0, stream>>>(q2h, k2h, Sbuf);
  softmax_rows<<<MF / 8, 256, 0, stream>>>(Sbuf, temporal_m, Pbuf, 1.0f, 0);
  wmma_apply<<<M1, 256, 0, stream>>>(Pbuf, H1T, H2, 1);    // H2 = h2 natural layout
  extract_diag<<<M1, 256, 0, stream>>>(H2, dh2);           // d2

  // ---- stage 3 (only diagonal q rows needed) ----
  wmma_linear<<<gBig, 256, 0, stream>>>(H2, wak, ak_b, k3h, nullptr, MF, HS, HS);
  wmma_linear<<<gSmall, 256, 0, stream>>>(dh2, waq, aq_b, q3dh, nullptr, M1, HS, HS);
  stage3_diag<<<M1, 256, 0, stream>>>(q3dh, k3h, H2, sasrec_m, d3h, scale);

  // ---- head ----
  build_cat<<<M1, 256, 0, stream>>>(dh1, dh2, d3h, catC);
  wmma_linear<<<gSmall, 256, 0, stream>>>(catC, wct, cat_b, nullptr, Af, M1, HS, 3 * HS);
  norm_rows<<<M1, 256, 0, stream>>>(Af, Anh);
  norm_rows<<<NM1, 256, 0, stream>>>(emb + HS, Bnh);       // emb[1:]

  dim3 gScore((M1 + 63) / 64, (NM1 + 127) / 128);          // 50 x 313
  wmma_linear<<<gScore, 256, 0, stream>>>(Anh, Bnh, nullptr, nullptr, scores_out,
                                          M1, NM1, HS);
}
